// CVRP_Decoder_56487409877524
// MI455X (gfx1250) — compile-verified, matrix operand order checked by reading, not run on hardware
//
#include <hip/hip_runtime.h>
#include <hip/hip_bf16.h>

// ---------------------------------------------------------------------------
// CVRP decoder forward, fused for gfx1250 (MI455X).
// All GEMMs via v_wmma_f32_16x16x32_bf16 (wave32 WMMA); transcendentals fp32.
// ---------------------------------------------------------------------------

#define B_ 64
#define P_ 512
#define N_ 2048
#define E_ 128

typedef __bf16 bf16_t;
typedef __attribute__((ext_vector_type(16))) __bf16 v16bf;
typedef __attribute__((ext_vector_type(8)))  __bf16 v8bf;
typedef __attribute__((ext_vector_type(8)))  float  v8f;

__device__ __forceinline__ v8f wmma_bf16(v16bf a, v16bf b, v8f c) {
    // (neg_a, A, neg_b, B, c_mod, C, reuse_a, reuse_b)
    return __builtin_amdgcn_wmma_f32_16x16x32_bf16(false, a, false, b,
                                                   (short)0, c, false, false);
}

__device__ __forceinline__ v8f vzero8() {
    v8f z = {0.f, 0.f, 0.f, 0.f, 0.f, 0.f, 0.f, 0.f};
    return z;
}

// Load 16 contiguous f32 (64B, aligned) and convert to a bf16 A/B fragment half.
__device__ __forceinline__ v16bf cvt_load16(const float* __restrict__ p) {
    float t[16];
    const float4* q = (const float4*)p;
#pragma unroll
    for (int i = 0; i < 4; ++i) {
        float4 x = q[i];
        t[4 * i + 0] = x.x; t[4 * i + 1] = x.y;
        t[4 * i + 2] = x.z; t[4 * i + 3] = x.w;
    }
    v16bf r;
#pragma unroll
    for (int i = 0; i < 16; ++i) r[i] = (__bf16)t[i];
    return r;
}

// ---------------------------------------------------------------------------
// Kernel 0: transpose-convert weights to bf16.  WT[e][k] = (bf16)W[k][e].
// Wq_last: only rows 0..127 go through the GEMM path (rows 128/129 are the
// rank-2 load/left update, applied in fp32 in the q kernel epilogue).
// ---------------------------------------------------------------------------
__global__ void convert_weights_kernel(const float* __restrict__ Wq1,
                                       const float* __restrict__ Wq2,
                                       const float* __restrict__ Wq_last,
                                       const float* __restrict__ Wk,
                                       const float* __restrict__ Wv,
                                       bf16_t* __restrict__ Wq1T,
                                       bf16_t* __restrict__ Wq2T,
                                       bf16_t* __restrict__ WqlT,
                                       bf16_t* __restrict__ WkT,
                                       bf16_t* __restrict__ WvT) {
    int idx = blockIdx.x * blockDim.x + threadIdx.x;
    if (idx >= E_ * E_) return;
    int e = idx / E_;
    int k = idx % E_;
    int src = k * E_ + e;
    Wq1T[idx] = (__bf16)Wq1[src];
    Wq2T[idx] = (__bf16)Wq2[src];
    WqlT[idx] = (__bf16)Wq_last[src];   // (E+2,E): rows 0..127, stride E
    WkT[idx]  = (__bf16)Wk[src];
    WvT[idx]  = (__bf16)Wv[src];
}

// ---------------------------------------------------------------------------
// Kernel 1: per (b, 16-row N-tile) wave:
//   k = nodes@Wk, v = nodes@Wv via WMMA, then ek = exp(k), ekv = ek*v.
//   Store ekT / ekvT transposed as (B, E, N) bf16 (B-operand-ready), and a
//   bf16 copy of encoded_nodes (B, N, E) for the score GEMM.
// ---------------------------------------------------------------------------
__global__ void prep_kv_kernel(const float* __restrict__ nodes,
                               const bf16_t* __restrict__ WkT,
                               const bf16_t* __restrict__ WvT,
                               bf16_t* __restrict__ nodes_bf,
                               bf16_t* __restrict__ ekT,
                               bf16_t* __restrict__ ekvT) {
    int gwave = (blockIdx.x * blockDim.x + threadIdx.x) >> 5;
    int lane  = threadIdx.x & 31;
    int b  = gwave / (N_ / 16);
    int nt = gwave % (N_ / 16);
    int n0 = nt * 16;
    int m  = lane & 15;     // row within tile (A) / column (B,C)
    int kh = lane >> 4;     // K-half selector

    // A fragments: rows of encoded_nodes, converted to bf16 (resident for all
    // 8 column-tiles).  Lane holds 16 contiguous K elements per fragment.
    const float* arow = nodes + ((size_t)b * N_ + n0 + m) * E_;
    v16bf afr[4];
#pragma unroll
    for (int kk = 0; kk < 4; ++kk) {
        int ko = kk * 32 + kh * 16;
        afr[kk] = cvt_load16(arow + ko);
        // bf16 copy of encoded_nodes (each lane covers its half of each row)
        bf16_t* dst = nodes_bf + ((size_t)b * N_ + n0 + m) * E_ + ko;
        *(v16bf*)dst = afr[kk];
    }

    for (int et = 0; et < 8; ++et) {           // 8 column tiles of E
        v8f ck = vzero8();
        v8f cv = vzero8();
#pragma unroll
        for (int kk = 0; kk < 4; ++kk) {
            int ko = kk * 32 + kh * 16;
            const bf16_t* bk = WkT + (size_t)(et * 16 + m) * E_ + ko;
            const bf16_t* bv = WvT + (size_t)(et * 16 + m) * E_ + ko;
            v16bf bfk = *(const v16bf*)bk;
            v16bf bfv = *(const v16bf*)bv;
            ck = wmma_bf16(afr[kk], bfk, ck);
            cv = wmma_bf16(afr[kk], bfv, cv);
        }
        // Epilogue: ek = exp(k); ekv = ek * v.  C layout: VGPR j, lanes 0-15
        // -> row M=j, lanes 16-31 -> M=j+8; column = m.  Store transposed:
        // ekT[b][e=et*16+m][n = n0 + 8*kh + j] -> 8 contiguous bf16 per lane.
        v8bf pek, pekv;
#pragma unroll
        for (int j = 0; j < 8; ++j) {
            float ek = __expf(ck[j]);
            pek[j]  = (__bf16)ek;
            pekv[j] = (__bf16)(ek * cv[j]);
        }
        size_t off = ((size_t)b * E_ + et * 16 + m) * N_ + n0 + 8 * kh;
        *(v8bf*)(ekT  + off) = pek;
        *(v8bf*)(ekvT + off) = pekv;
    }
}

// ---------------------------------------------------------------------------
// Kernel 2: q = q1@Wq1 + q2@Wq2 + last@Wq_last[:128] + load*w128 + left*w129,
// then sigmoid; stored fp32 (B,P,E).  One wave per 16 P-rows.
// ---------------------------------------------------------------------------
__global__ void q_sigmoid_kernel(const float* __restrict__ enc_last,
                                 const float* __restrict__ loadv,
                                 const float* __restrict__ leftv,
                                 const float* __restrict__ enc_q1,
                                 const float* __restrict__ enc_q2,
                                 const bf16_t* __restrict__ Wq1T,
                                 const bf16_t* __restrict__ Wq2T,
                                 const bf16_t* __restrict__ WqlT,
                                 const float* __restrict__ Wq_last,
                                 float* __restrict__ sig_q) {
    int gwave = (blockIdx.x * blockDim.x + threadIdx.x) >> 5;
    int lane  = threadIdx.x & 31;
    int b  = gwave / (P_ / 16);
    int pt = gwave % (P_ / 16);
    int p0 = pt * 16;
    int m  = lane & 15;
    int kh = lane >> 4;

    const size_t rowbase = ((size_t)b * P_ + p0 + m) * E_;
    for (int et = 0; et < 8; ++et) {
        v8f c = vzero8();
#pragma unroll
        for (int kk = 0; kk < 4; ++kk) {
            int ko = kk * 32 + kh * 16;
            size_t woff = (size_t)(et * 16 + m) * E_ + ko;
            c = wmma_bf16(cvt_load16(enc_q1 + rowbase + ko),
                          *(const v16bf*)(Wq1T + woff), c);
            c = wmma_bf16(cvt_load16(enc_q2 + rowbase + ko),
                          *(const v16bf*)(Wq2T + woff), c);
            c = wmma_bf16(cvt_load16(enc_last + rowbase + ko),
                          *(const v16bf*)(WqlT + woff), c);
        }
        int e = et * 16 + m;
        float lw0 = Wq_last[128 * E_ + e];
        float lw1 = Wq_last[129 * E_ + e];
#pragma unroll
        for (int j = 0; j < 8; ++j) {
            int p = p0 + j + 8 * kh;
            float qv = c[j] + loadv[b * P_ + p] * lw0 + leftv[b * P_ + p] * lw1;
            float sg = 1.f / (1.f + __expf(-qv));
            sig_q[((size_t)b * P_ + p) * E_ + e] = sg;
        }
    }
}

// ---------------------------------------------------------------------------
// Kernel 3: fused AFT + score + unnormalized softmax.
// Block = 128 threads (4 waves), 32 P-rows of one batch.
// Phase 1: stream N in 128-chunks; cooperative e_bias -> LDS (bf16);
//          waves split (row-tile x E-half); WMMA numer/denom accumulation.
// Phase 2: aft = sigmoid_q * nan_to_num(numer)/ (nan_to_num(denom)+1e-20),
//          staged to LDS as bf16 (A-operand of score GEMM).
// Phase 3: waves split (row-tile x N-half); score = aft @ nodes^T via WMMA;
//          epilogue applies /sqrt(E), distance bias, 10*tanh, mask, exp;
//          writes unnormalized probs + per-row sums (shfl + ds_add_f32).
// ---------------------------------------------------------------------------
#define NCHUNK 128

__global__ void aft_score_kernel(const float* __restrict__ cur_dist,
                                 const float* __restrict__ ninf,
                                 const float* __restrict__ sig_q,
                                 const bf16_t* __restrict__ ekT,
                                 const bf16_t* __restrict__ ekvT,
                                 const bf16_t* __restrict__ nodes_bf,
                                 const float* __restrict__ log_scale,
                                 const float* __restrict__ aft_alpha,
                                 const float* __restrict__ probs_alpha,
                                 float* __restrict__ probs_out,
                                 float* __restrict__ rowsum_out) {
    __shared__ __align__(32) bf16_t ebias_lds[32][NCHUNK];
    __shared__ __align__(32) bf16_t aft_lds[32][E_];
    __shared__ float rowsum_lds[32];

    int tid  = threadIdx.x;          // 0..127
    int wave = tid >> 5;             // 0..3
    int lane = tid & 31;
    int m  = lane & 15;
    int kh = lane >> 4;

    int b  = blockIdx.x / (P_ / 32);
    int pt = blockIdx.x % (P_ / 32);
    int p0 = pt * 32;
    const size_t bp0 = (size_t)b * P_ + p0;

    float ls    = log_scale[0];
    float coefA = -ls * aft_alpha[0];     // alpha_bias = coefA*dist + mask
    float coefS = -ls * probs_alpha[0];   // score bias = coefS*dist

    if (tid < 32) rowsum_lds[tid] = 0.f;

    int rt = wave >> 1;   // row tile (16 rows) for phase 1 & 3
    int eh = wave & 1;    // E-half for phase 1

    v8f numer[4], denom[4];
#pragma unroll
    for (int et = 0; et < 4; ++et) { numer[et] = vzero8(); denom[et] = vzero8(); }

    for (int n0 = 0; n0 < N_; n0 += NCHUNK) {
        // --- cooperative e_bias chunk: 32 rows x 128 cols, 32 vals/thread ---
        {
            int row = tid >> 2;
            int cb  = (tid & 3) * 32;
            const float* dsrc = cur_dist + (bp0 + row) * N_ + n0 + cb;
            const float* msrc = ninf     + (bp0 + row) * N_ + n0 + cb;
#pragma unroll
            for (int i = 0; i < 32; i += 4) {
                float4 d  = *(const float4*)(dsrc + i);
                float4 mk = *(const float4*)(msrc + i);
                ebias_lds[row][cb + i + 0] = (__bf16)__expf(coefA * d.x + mk.x);
                ebias_lds[row][cb + i + 1] = (__bf16)__expf(coefA * d.y + mk.y);
                ebias_lds[row][cb + i + 2] = (__bf16)__expf(coefA * d.z + mk.z);
                ebias_lds[row][cb + i + 3] = (__bf16)__expf(coefA * d.w + mk.w);
            }
        }
        __syncthreads();

        // --- WMMA accumulation over this chunk ---
#pragma unroll
        for (int kc = 0; kc < NCHUNK / 32; ++kc) {
            v16bf afr = *(const v16bf*)&ebias_lds[rt * 16 + m][kc * 32 + kh * 16];
#pragma unroll
            for (int et = 0; et < 4; ++et) {
                size_t boff = ((size_t)b * E_ + eh * 64 + et * 16 + m) * N_
                            + n0 + kc * 32 + kh * 16;
                v16bf bv = *(const v16bf*)(ekvT + boff);
                v16bf bk = *(const v16bf*)(ekT  + boff);
                numer[et] = wmma_bf16(afr, bv, numer[et]);
                denom[et] = wmma_bf16(afr, bk, denom[et]);
            }
        }
        __syncthreads();   // protect ebias_lds before next chunk overwrite
    }

    // --- phase 2: aft epilogue in registers -> LDS (bf16) ---
#pragma unroll
    for (int et = 0; et < 4; ++et) {
#pragma unroll
        for (int j = 0; j < 8; ++j) {
            int prow = rt * 16 + j + 8 * kh;
            int e    = eh * 64 + et * 16 + m;
            float nn = numer[et][j]; nn = (nn == nn) ? nn : 0.f;  // nan_to_num
            float dd = denom[et][j]; dd = (dd == dd) ? dd : 0.f;
            float w  = nn / (dd + 1e-20f);
            float sg = sig_q[(bp0 + prow) * E_ + e];
            aft_lds[prow][e] = (__bf16)(sg * w);
        }
    }
    __syncthreads();

    // --- phase 3: score GEMM + softmax numerator ---
    int nh = wave & 1;   // N-half
    v16bf afr3[4];
#pragma unroll
    for (int kk = 0; kk < 4; ++kk)
        afr3[kk] = *(const v16bf*)&aft_lds[rt * 16 + m][kk * 32 + kh * 16];

    float rs[8];
#pragma unroll
    for (int j = 0; j < 8; ++j) rs[j] = 0.f;

    const float inv_sqrtE = 0.08838834764831845f;   // 1/sqrt(128)
    for (int nb = nh * (N_ / 2); nb < (nh + 1) * (N_ / 2); nb += 16) {
        v8f c = vzero8();
#pragma unroll
        for (int kk = 0; kk < 4; ++kk) {
            const bf16_t* bp = nodes_bf
                + ((size_t)b * N_ + nb + m) * E_ + kk * 32 + kh * 16;
            c = wmma_bf16(afr3[kk], *(const v16bf*)bp, c);
        }
#pragma unroll
        for (int j = 0; j < 8; ++j) {
            int prow = rt * 16 + j + 8 * kh;
            int n    = nb + m;
            size_t off = (bp0 + prow) * N_ + n;
            float d  = cur_dist[off];
            float mk = ninf[off];
            float s  = c[j] * inv_sqrtE + coefS * d;
            float t  = 10.f * tanhf(s);
            float e  = __expf(t + mk);
            rs[j] += e;
            probs_out[off] = e;   // unnormalized; rescaled in kernel 4
        }
    }

    // per-row sum: reduce across the 16-lane column groups, then LDS atomics
#pragma unroll
    for (int j = 0; j < 8; ++j) {
        float v = rs[j];
        v += __shfl_xor(v, 1, 32);
        v += __shfl_xor(v, 2, 32);
        v += __shfl_xor(v, 4, 32);
        v += __shfl_xor(v, 8, 32);
        if (m == 0) atomicAdd(&rowsum_lds[rt * 16 + j + 8 * kh], v);
    }
    __syncthreads();
    if (tid < 32) rowsum_out[bp0 + tid] = rowsum_lds[tid];
}

// ---------------------------------------------------------------------------
// Kernel 4: probs /= rowsum  (float4 vectorized).
// ---------------------------------------------------------------------------
__global__ void rescale_kernel(float* __restrict__ probs,
                               const float* __restrict__ rowsum) {
    size_t idx = (size_t)blockIdx.x * blockDim.x + threadIdx.x;
    size_t total = (size_t)B_ * P_ * (N_ / 4);
    if (idx >= total) return;
    size_t row = idx / (N_ / 4);
    float inv = 1.0f / rowsum[row];
    float4* p4 = (float4*)probs + idx;
    float4 v = *p4;
    v.x *= inv; v.y *= inv; v.z *= inv; v.w *= inv;
    *p4 = v;
}

// ---------------------------------------------------------------------------
extern "C" void kernel_launch(void* const* d_in, const int* in_sizes, int n_in,
                              void* d_out, int out_size, void* d_ws, size_t ws_size,
                              hipStream_t stream) {
    (void)in_sizes; (void)n_in; (void)out_size; (void)ws_size;

    const float* enc_last   = (const float*)d_in[0];   // (B,P,E)
    const float* loadv      = (const float*)d_in[1];   // (B,P)
    const float* leftv      = (const float*)d_in[2];   // (B,P)
    const float* cur_dist   = (const float*)d_in[3];   // (B,P,N)
    const float* log_scale  = (const float*)d_in[4];   // (1,)
    const float* ninf_mask  = (const float*)d_in[5];   // (B,P,N)
    const float* enc_nodes  = (const float*)d_in[6];   // (B,N,E)
    const float* enc_q1     = (const float*)d_in[7];   // (B,P,E)
    const float* enc_q2     = (const float*)d_in[8];   // (B,P,E)
    const float* Wq1        = (const float*)d_in[9];   // (E,E)
    const float* Wq2        = (const float*)d_in[10];  // (E,E)
    const float* Wq_last    = (const float*)d_in[11];  // (E+2,E)
    const float* Wk         = (const float*)d_in[12];  // (E,E)
    const float* Wv         = (const float*)d_in[13];  // (E,E)
    const float* aft_alpha  = (const float*)d_in[14];  // (1,)
    const float* probs_alp  = (const float*)d_in[15];  // (1,)
    float* probs = (float*)d_out;                      // (B,P,N)

    // Workspace carve-up (all offsets 256B-aligned).
    char* ws = (char*)d_ws;
    size_t off = 0;
    bf16_t* nodes_bf = (bf16_t*)(ws + off); off += (size_t)B_ * N_ * E_ * 2;
    bf16_t* ekT      = (bf16_t*)(ws + off); off += (size_t)B_ * E_ * N_ * 2;
    bf16_t* ekvT     = (bf16_t*)(ws + off); off += (size_t)B_ * E_ * N_ * 2;
    float*  sig_q    = (float*) (ws + off); off += (size_t)B_ * P_ * E_ * 4;
    bf16_t* WkT      = (bf16_t*)(ws + off); off += (size_t)E_ * E_ * 2;
    bf16_t* WvT      = (bf16_t*)(ws + off); off += (size_t)E_ * E_ * 2;
    bf16_t* Wq1T     = (bf16_t*)(ws + off); off += (size_t)E_ * E_ * 2;
    bf16_t* Wq2T     = (bf16_t*)(ws + off); off += (size_t)E_ * E_ * 2;
    bf16_t* WqlT     = (bf16_t*)(ws + off); off += (size_t)E_ * E_ * 2;
    float*  rowsum   = (float*) (ws + off); off += (size_t)B_ * P_ * 4;

    // K0: weight transpose-convert (16384 elems)
    convert_weights_kernel<<<(E_ * E_ + 255) / 256, 256, 0, stream>>>(
        Wq1, Wq2, Wq_last, Wk, Wv, Wq1T, Wq2T, WqlT, WkT, WvT);

    // K1: k/v projection + exp fusion.  B*(N/16)=8192 waves, 8 waves/block.
    prep_kv_kernel<<<(B_ * (N_ / 16)) / 8, 256, 0, stream>>>(
        enc_nodes, WkT, WvT, nodes_bf, ekT, ekvT);

    // K2: q + sigmoid.  B*(P/16)=2048 waves, 8 waves/block.
    q_sigmoid_kernel<<<(B_ * (P_ / 16)) / 8, 256, 0, stream>>>(
        enc_last, loadv, leftv, enc_q1, enc_q2, Wq1T, Wq2T, WqlT, Wq_last, sig_q);

    // K3: fused AFT + score + unnormalized softmax.  B*(P/32)=1024 blocks.
    aft_score_kernel<<<B_ * (P_ / 32), 128, 0, stream>>>(
        cur_dist, ninf_mask, sig_q, ekT, ekvT, nodes_bf,
        log_scale, aft_alpha, probs_alp, probs, rowsum);

    // K4: normalize.
    rescale_kernel<<<(int)(((size_t)B_ * P_ * (N_ / 4) + 255) / 256), 256, 0, stream>>>(
        probs, rowsum);
}
